// MARE_89361089560620
// MI455X (gfx1250) — matrix operand order repaired and checked by reading, not compiled, not run on hardware
//
#include <hip/hip_runtime.h>
#include <hip/hip_bf16.h>

typedef _Float16 half_t;
typedef __attribute__((ext_vector_type(8)))  _Float16 v8h;
typedef __attribute__((ext_vector_type(16))) _Float16 v16h;
typedef __attribute__((ext_vector_type(8)))  float    v8f;

#define T_TOT   1024
#define SEN     120
#define EMB     60
#define EMBP    64
#define DIMC    230
#define DIMCP   240
#define DIMR    176
#define NUMIN   128
#define SOUT    118   // SEN - 2

// ---------------------------------------------------------------- utility
__global__ void amnre_zero(float* o, int n) {
    int i = blockIdx.x * 256 + threadIdx.x;
    if (i < n) o[i] = 0.f;
}

__global__ void amnre_starts(const int* lA, const int* lB, int* sA, int* sB) {
    if (threadIdx.x == 0 && blockIdx.x == 0) {
        int a = 0, b = 0;
        for (int i = 0; i < NUMIN; ++i) {
            sA[i] = a; a += lA[i];
            sB[i] = b; b += lB[i];
        }
    }
}

// ------------------------------------------------- weight prep: [3][64][240] f16
__global__ void amnre_prepw(const float* __restrict__ convW, half_t* __restrict__ Wp) {
    int idx = blockIdx.x * 256 + threadIdx.x;
    if (idx >= 3 * EMBP * DIMCP) return;
    int n  = idx % DIMCP;
    int fe = idx / DIMCP;
    int e  = fe % EMBP;
    int f  = fe / EMBP;
    float v = 0.f;
    if (n < DIMC && e < EMB) v = convW[((size_t)n * EMB + e) * 3 + f];
    Wp[idx] = (half_t)v;
}

// ------------------------------------------------- embed gather -> f16 [T,120,64]
__global__ void amnre_embed(const int* __restrict__ words, const int* __restrict__ pos1,
                            const int* __restrict__ pos2,
                            const float* __restrict__ wemb, const float* __restrict__ p1,
                            const float* __restrict__ p2, half_t* __restrict__ xout) {
    int idx = blockIdx.x * 256 + threadIdx.x;
    if (idx >= T_TOT * SEN * EMBP) return;
    int d  = idx & (EMBP - 1);
    int ts = idx >> 6;                 // t*120 + s
    float v = 0.f;
    if      (d < 50) v = wemb[(size_t)words[ts] * 50 + d];
    else if (d < 55) v = p1[pos1[ts] * 5 + (d - 50)];
    else if (d < 60) v = p2[pos2[ts] * 5 + (d - 55)];
    xout[idx] = (half_t)v;
}

// ------------------------------------------------- conv-as-WMMA + maxpool + tanh
// grid = T sentences, block = 256 (8 waves). Wave w owns M-tile w (rows w*16..w*16+15
// of the 118-row im2col), and all 15 N-tiles (240 >= 230 channels).
// Per filter tap f: stage a [240 ch][64 k] transposed weight slice in LDS, load both
// k-step A fragments once, then run 30 software-pipelined (double-buffered B) WMMAs.
__launch_bounds__(256, 1)
__global__ void amnre_conv(const half_t* __restrict__ X, const half_t* __restrict__ Wp,
                           const float* __restrict__ bias, float* __restrict__ out) {
    __shared__ half_t xs[132 * EMBP];        // sentence, zero-padded to 132 rows (16.9 KB)
    __shared__ half_t wsB[DIMCP * EMBP];     // transposed weight slice [n][k]   (30.7 KB)
    __shared__ float  pool[8][DIMCP];        //                                   (7.7 KB)

    int t = blockIdx.x, tid = threadIdx.x;

    // stage sentence (120*64 halves = 3840 dwords), zero pad rows 120..131
    const unsigned int* src = (const unsigned int*)(X + (size_t)t * SEN * EMBP);
    unsigned int* dx = (unsigned int*)xs;
    for (int i = tid; i < 3840; i += 256) dx[i] = src[i];
    for (int i = 3840 + tid; i < 132 * EMBP / 2; i += 256) dx[i] = 0u;

    int wave = tid >> 5, lane = tid & 31;
    int hb = lane >> 4, lm = lane & 15;
    int m0 = wave * 16;

    v8f acc[15];
    const v8f vz = {0.f, 0.f, 0.f, 0.f, 0.f, 0.f, 0.f, 0.f};
#pragma unroll
    for (int nt = 0; nt < 15; ++nt) acc[nt] = vz;

    for (int f = 0; f < 3; ++f) {
        __syncthreads();   // also covers xs staging on first pass
        // stage full 64-wide W slice transposed: wsB[n*64 + k] = Wp[(f*64 + k)*240 + n]
        for (int i = tid; i < DIMCP * EMBP; i += 256) {
            int k = i / DIMCP, n = i % DIMCP;
            wsB[n * EMBP + k] = Wp[(f * EMBP + k) * DIMCP + n];
        }
        __syncthreads();

        // A fragments for both 32-wide k-steps (this code's contiguous k order, same as B:
        // element j -> k = half*16 + j within the k-step)
        const half_t* xrow = &xs[(m0 + lm + f) * EMBP + hb * 16];
        v8h lo0 = *(const v8h*)(xrow);
        v8h hi0 = *(const v8h*)(xrow + 8);
        v8h lo1 = *(const v8h*)(xrow + 32);
        v8h hi1 = *(const v8h*)(xrow + 40);
        v16h a0 = __builtin_shufflevector(lo0, hi0,
            0, 1, 2, 3, 4, 5, 6, 7, 8, 9, 10, 11, 12, 13, 14, 15);
        v16h a1 = __builtin_shufflevector(lo1, hi1,
            0, 1, 2, 3, 4, 5, 6, 7, 8, 9, 10, 11, 12, 13, 14, 15);

        // B fragment base: column n = nt*16+lm, element j -> k = half*16 + j
        // b(nt,kk) = 32B contiguous at wsB[(nt*16+lm)*64 + kk*32 + hb*16]
        const half_t* bbase = &wsB[lm * EMBP + hb * 16];

        // 30 WMMAs, kk-outer so consecutive WMMAs hit different accumulators;
        // double-buffered B fragments so ds_load of the next overlaps the current WMMA.
        v16h bcur = *(const v16h*)(bbase);
#pragma unroll
        for (int u = 0; u < 30; ++u) {
            int nt = (u < 15) ? u : (u - 15);
            int kk = (u < 15) ? 0 : 1;
            v16h bnext = bcur;
            if (u < 29) {
                int un = u + 1;
                int ntn = (un < 15) ? un : (un - 15);
                int kkn = (un < 15) ? 0 : 1;
                bnext = *(const v16h*)(bbase + ntn * 1024 + kkn * 32);
            }
            acc[nt] = __builtin_amdgcn_wmma_f32_16x16x32_f16(
                false, (kk ? a1 : a0), false, bcur, (short)0, acc[nt], false, false);
            bcur = bnext;
        }
    }

    // fused maxpool over the sequence positions this wave owns
#pragma unroll
    for (int nt = 0; nt < 15; ++nt) {
        float vmax = -3.0e38f;
#pragma unroll
        for (int i = 0; i < 8; ++i) {
            int s = m0 + i + hb * 8;           // D layout: VGPR i -> M = i + 8*half
            if (s < SOUT) vmax = fmaxf(vmax, acc[nt][i]);
        }
        vmax = fmaxf(vmax, __shfl_xor(vmax, 16, 32));
        if (hb == 0) pool[wave][nt * 16 + lm] = vmax;
    }
    __syncthreads();
    if (tid < DIMC) {
        float m = pool[0][tid];
#pragma unroll
        for (int w = 1; w < 8; ++w) m = fmaxf(m, pool[w][tid]);
        out[(size_t)t * DIMC + tid] = tanhf(m + bias[tid]);
    }
}

// ------------------------------------------------- scores[k,t] = rel[k] . inp[t]
__global__ void amnre_scores(const float* __restrict__ rel, const float* __restrict__ inp,
                             float* __restrict__ sc) {
    __shared__ float sIn[32 * 232];
    int t0 = blockIdx.x * 32, tid = threadIdx.x;
    for (int i = tid; i < 32 * DIMC; i += 256)
        sIn[(i / DIMC) * 232 + (i % DIMC)] = inp[(size_t)t0 * DIMC + i];
    __syncthreads();
    for (int o = tid; o < DIMR * 32; o += 256) {
        int k = o >> 5, tt = o & 31;
        const float* rk = rel + (size_t)k * DIMC;
        const float* xi = sIn + tt * 232;
        float acc = 0.f;
        for (int d = 0; d < DIMC; ++d) acc += rk[d] * xi[d];
        sc[(size_t)k * T_TOT + t0 + tt] = acc;
    }
}

// ------------------------------------------------- fused segmented attend + head
// grid = (176 relations, 128 bags), block = 256
__global__ void amnre_head(const float* __restrict__ scA, const float* __restrict__ scB,
                           const int* __restrict__ rA, const int* __restrict__ rB,
                           const int* __restrict__ stA, const int* __restrict__ stB,
                           const int* __restrict__ lA, const int* __restrict__ lB,
                           const float* __restrict__ inpA, const float* __restrict__ inpB,
                           const float* __restrict__ rel, const float* __restrict__ MW,
                           const float* __restrict__ Mb, const int* __restrict__ re_mask,
                           float* __restrict__ out, int multi) {
    __shared__ float wbuf[2048];
    __shared__ float Svec[232];
    __shared__ float larr[192];
    __shared__ float red[256];

    int j = blockIdx.x, n = blockIdx.y, tid = threadIdx.x;
    int sA = stA[n], nA = lA[n];
    int sB = 0, nB = 0;
    if (multi) { sB = stB[n]; nB = lB[n]; }
    int tot = nA + nB;

    // gather attention logits for this (bag, relation)
    for (int i = tid; i < tot; i += 256) {
        float a;
        if (i < nA) { int t = sA + i;        a = scA[(size_t)rA[j * T_TOT + t] * T_TOT + t]; }
        else        { int t = sB + (i - nA); a = scB[(size_t)rB[j * T_TOT + t] * T_TOT + t]; }
        wbuf[i] = a;
    }
    __syncthreads();

    // max
    float v = -3.0e38f;
    for (int i = tid; i < tot; i += 256) v = fmaxf(v, wbuf[i]);
    red[tid] = v; __syncthreads();
    for (int s = 128; s; s >>= 1) { if (tid < s) red[tid] = fmaxf(red[tid], red[tid + s]); __syncthreads(); }
    float mx = red[0]; __syncthreads();

    // exp + sum
    float sum = 0.f;
    for (int i = tid; i < tot; i += 256) { float e = __expf(wbuf[i] - mx); wbuf[i] = e; sum += e; }
    red[tid] = sum; __syncthreads();
    for (int s = 128; s; s >>= 1) { if (tid < s) red[tid] += red[tid + s]; __syncthreads(); }
    float invz = 1.0f / red[0]; __syncthreads();

    // pooled sentence vector S[n,j,:]
    if (tid < DIMC) {
        float acc = 0.f;
        for (int i = 0; i < nA; ++i) acc += wbuf[i]      * inpA[(size_t)(sA + i) * DIMC + tid];
        for (int i = 0; i < nB; ++i) acc += wbuf[nA + i] * inpB[(size_t)(sB + i) * DIMC + tid];
        Svec[tid] = acc * invz;
    }
    __syncthreads();

    // diag = rel[r[j, start]] . S
    int rr = rA[j * T_TOT + sA];
    float pd = 0.f;
    for (int d = tid; d < DIMC; d += 256) pd += rel[(size_t)rr * DIMC + d] * Svec[d];
    red[tid] = pd; __syncthreads();
    for (int s = 128; s; s >>= 1) { if (tid < s) red[tid] += red[tid + s]; __syncthreads(); }
    float diag = red[0]; __syncthreads();

    // logits over 176 classes
    if (tid < DIMR) {
        float lg = Mb[tid] + diag;
        const float* mw = MW + (size_t)tid * DIMC;
        for (int d = 0; d < DIMC; ++d) lg += mw[d] * Svec[d];
        larr[tid] = lg;
    }
    __syncthreads();

    // log-softmax + masked gather
    float lv = (tid < DIMR) ? larr[tid] : -3.0e38f;
    red[tid] = lv; __syncthreads();
    for (int s = 128; s; s >>= 1) { if (tid < s) red[tid] = fmaxf(red[tid], red[tid + s]); __syncthreads(); }
    float lmx = red[0]; __syncthreads();
    float le = (tid < DIMR) ? __expf(larr[tid] - lmx) : 0.f;
    red[tid] = le; __syncthreads();
    for (int s = 128; s; s >>= 1) { if (tid < s) red[tid] += red[tid + s]; __syncthreads(); }
    float lz = red[0]; __syncthreads();
    if (tid == 0) {
        int kk = re_mask[n * DIMR + j];
        out[n * DIMR + j] += larr[kk] - lmx - logf(lz);
    }
}

// ================================================================ host side
struct AmnrePtrs {
    const float *wemb[4], *p1[4], *p2[4], *convW[4], *convb[4]; // share_en, share_zh, mono_en, mono_zh
    const float *rel[3], *MW[3], *Mb[3];                        // enRE, zhRE, multiRE
    const int *words[2], *pos1[2], *pos2[2], *r[2], *len[2];    // En, Zh
    const int *re_mask;
};

static void amnre_resolve(void* const* d_in, const int* in_sizes, AmnrePtrs& P) {
    auto F = [&](int i) { return (const float*)d_in[i]; };
    auto I = [&](int i) { return (const int*)d_in[i]; };

    if (in_sizes[0] == 128) {
        // fully sorted (jax tree_flatten of whole input dict, dicts alphabetical)
        P.len[0] = I(0); P.len[1] = I(1);
        P.MW[0] = F(2);  P.Mb[0] = F(3);  P.rel[0] = F(4);
        int encIdx[4] = {18, 23, 5, 10};  // share_en, share_zh, mono_en, mono_zh
        for (int e = 0; e < 4; ++e) {
            int b = encIdx[e];
            P.convW[e] = F(b); P.convb[e] = F(b + 1);
            P.p1[e] = F(b + 2); P.p2[e] = F(b + 3); P.wemb[e] = F(b + 4);
        }
        P.MW[2] = F(15); P.Mb[2] = F(16); P.rel[2] = F(17);
        P.MW[1] = F(28); P.Mb[1] = F(29); P.rel[1] = F(30);
        P.pos1[0] = I(31); P.pos1[1] = I(32); P.pos2[0] = I(33); P.pos2[1] = I(34);
        P.r[0] = I(35); P.r[1] = I(36); P.re_mask = I(37);
        P.words[0] = I(38); P.words[1] = I(39);
        return;
    }

    bool paramsSorted = (in_sizes[0] == 40480); // enRE.M_W first
    if (paramsSorted) {
        P.MW[0] = F(0); P.Mb[0] = F(1); P.rel[0] = F(2);
        int encIdx[4] = {16, 21, 3, 8};   // share_en, share_zh, mono_en, mono_zh
        for (int e = 0; e < 4; ++e) {
            int b = encIdx[e];
            P.convW[e] = F(b); P.convb[e] = F(b + 1);
            P.p1[e] = F(b + 2); P.p2[e] = F(b + 3); P.wemb[e] = F(b + 4);
        }
        P.MW[2] = F(13); P.Mb[2] = F(14); P.rel[2] = F(15);
        P.MW[1] = F(26); P.Mb[1] = F(27); P.rel[1] = F(28);
    } else {
        // full insertion order: 4 encoders {wemb,p1,p2,convW,convb}, 3 RE {rel,MW,Mb}
        for (int e = 0; e < 4; ++e) {
            int b = e * 5;
            P.wemb[e] = F(b); P.p1[e] = F(b + 1); P.p2[e] = F(b + 2);
            P.convW[e] = F(b + 3); P.convb[e] = F(b + 4);
        }
        for (int h = 0; h < 3; ++h) {
            int b = 20 + h * 3;
            P.rel[h] = F(b); P.MW[h] = F(b + 1); P.Mb[h] = F(b + 2);
        }
    }
    P.words[0] = I(29); P.pos1[0] = I(30); P.pos2[0] = I(31);
    P.r[0] = I(32); P.len[0] = I(33);
    P.words[1] = I(34); P.pos1[1] = I(35); P.pos2[1] = I(36);
    P.r[1] = I(37); P.len[1] = I(38);
    P.re_mask = I(39);
}

extern "C" void kernel_launch(void* const* d_in, const int* in_sizes, int n_in,
                              void* d_out, int out_size, void* d_ws, size_t ws_size,
                              hipStream_t stream) {
    AmnrePtrs P;
    amnre_resolve(d_in, in_sizes, P);

    char* ws = (char*)d_ws;
    const size_t OFF_X   = 0;                              // 1024*120*64*2 = 15,728,640
    const size_t OFF_WP  = OFF_X + (size_t)T_TOT * SEN * EMBP * 2;
    const size_t SZ_WP   = (size_t)3 * EMBP * DIMCP * 2;   // 92,160
    const size_t OFF_ENC = OFF_WP + SZ_WP;
    const size_t SZ_ENC  = (size_t)T_TOT * DIMC * 4;       // 942,080
    const size_t OFF_SC  = OFF_ENC + 4 * SZ_ENC;
    const size_t SZ_SC   = (size_t)DIMR * T_TOT * 4;       // 720,896
    const size_t OFF_ST  = OFF_SC + 4 * SZ_SC;

    half_t* xbuf = (half_t*)(ws + OFF_X);
    half_t* wpad = (half_t*)(ws + OFF_WP);
    float*  enc[4];
    float*  sc[4];
    for (int i = 0; i < 4; ++i) enc[i] = (float*)(ws + OFF_ENC + i * SZ_ENC);
    for (int i = 0; i < 4; ++i) sc[i]  = (float*)(ws + OFF_SC + i * SZ_SC);
    int* stE = (int*)(ws + OFF_ST);
    int* stZ = stE + 128;
    float* out = (float*)d_out;

    amnre_zero<<<(NUMIN * DIMR + 255) / 256, 256, 0, stream>>>(out, NUMIN * DIMR);
    amnre_starts<<<1, 32, 0, stream>>>(P.len[0], P.len[1], stE, stZ);

    // encoders: 0=share_en, 1=share_zh, 2=mono_en, 3=mono_zh
    const int langOf[4] = {0, 1, 0, 1};
    const int nEmb = T_TOT * SEN * EMBP;
    for (int e = 0; e < 4; ++e) {
        int lg = langOf[e];
        amnre_prepw<<<(3 * EMBP * DIMCP + 255) / 256, 256, 0, stream>>>(P.convW[e], wpad);
        amnre_embed<<<(nEmb + 255) / 256, 256, 0, stream>>>(
            P.words[lg], P.pos1[lg], P.pos2[lg], P.wemb[e], P.p1[e], P.p2[e], xbuf);
        amnre_conv<<<T_TOT, 256, 0, stream>>>(xbuf, wpad, P.convb[e], enc[e]);
    }

    // score tables: mono_en(enRE), mono_zh(zhRE), share_en(multiRE), share_zh(multiRE)
    amnre_scores<<<T_TOT / 32, 256, 0, stream>>>(P.rel[0], enc[2], sc[0]);
    amnre_scores<<<T_TOT / 32, 256, 0, stream>>>(P.rel[1], enc[3], sc[1]);
    amnre_scores<<<T_TOT / 32, 256, 0, stream>>>(P.rel[2], enc[0], sc[2]);
    amnre_scores<<<T_TOT / 32, 256, 0, stream>>>(P.rel[2], enc[1], sc[3]);

    dim3 hg(DIMR, NUMIN);
    amnre_head<<<hg, 256, 0, stream>>>(sc[0], nullptr, P.r[0], nullptr, stE, nullptr,
                                       P.len[0], nullptr, enc[2], nullptr,
                                       P.rel[0], P.MW[0], P.Mb[0], P.re_mask, out, 0);
    amnre_head<<<hg, 256, 0, stream>>>(sc[1], nullptr, P.r[1], nullptr, stZ, nullptr,
                                       P.len[1], nullptr, enc[3], nullptr,
                                       P.rel[1], P.MW[1], P.Mb[1], P.re_mask, out, 0);
    amnre_head<<<hg, 256, 0, stream>>>(sc[2], sc[3], P.r[0], P.r[1], stE, stZ,
                                       P.len[0], P.len[1], enc[0], enc[1],
                                       P.rel[2], P.MW[2], P.Mb[2], P.re_mask, out, 1);
}
